// RegionProposalNetwork_55405078119174
// MI455X (gfx1250) — matrix operand order repaired and checked by reading, not compiled, not run on hardware
//
#include <hip/hip_runtime.h>

// ---------------- problem constants ----------------
#define BATCH 16
#define HF    24
#define WF    40
#define CIN   1024
#define CMID  512
#define NA    9
#define PPI   (HF*WF)          // 960 pixels per image
#define MTOT  (BATCH*PPI)      // 15360 GEMM rows
#define KTOT  (9*CIN)          // 9216 GEMM K
#define NANCH (PPI*NA)         // 8640 anchors per image
#define PRE_TOPK 6000
#define POST_N   300
#define NMS_T    0.7f

// conv GEMM tiling
#define KSTEPS   288           // 9216 / 32
#define CHUNK_K  128           // K values per LDS chunk (4 WMMA steps)
#define NCHUNK   72            // 9216 / 128
#define BCOLS    64            // N columns per block (shared via LDS)
#define BSTRIDE  136           // LDS row stride in elems: 128 + 8 pad -> 272B (16B-aligned, bank-spread)
#define MBLK     256           // M rows per block (8 waves x 32)

typedef __attribute__((ext_vector_type(16))) __bf16 v16bf;
typedef __attribute__((ext_vector_type(8)))  float  v8f;

#if defined(__AMDGCN__) && __has_builtin(__builtin_amdgcn_global_load_async_to_lds_b128)
#define HAVE_ASYNC_LDS 1
// builtin signature (from compiler diagnostic): param0 = AS(1) int-vec16*, non-const
typedef int v4i_raw __attribute__((vector_size(16)));
typedef __attribute__((address_space(1))) v4i_raw gv4;   // global
typedef __attribute__((address_space(3))) v4i_raw lv4;   // LDS
#else
#define HAVE_ASYNC_LDS 0
#endif

// scheduling pipeline control: masks per LLVM AMDGPU sched_group_barrier
//   0x008 = MFMA/WMMA, 0x020 = VMEM read, 0x100 = DS read
#if defined(__AMDGCN__) && __has_builtin(__builtin_amdgcn_sched_group_barrier)
#define SGB(mask, n) __builtin_amdgcn_sched_group_barrier(mask, n, 0)
#else
#define SGB(mask, n)
#endif

__device__ __forceinline__ __bf16 f2bf(float f) {
  unsigned u = __float_as_uint(f);
  u = (u + 0x7FFFu + ((u >> 16) & 1u)) >> 16;   // round-to-nearest-even
  unsigned short s = (unsigned short)u;
  __bf16 r; __builtin_memcpy(&r, &s, 2); return r;
}

// ---------------- kernel 1: pad + f32->bf16 activations ----------------
// xpad layout: [B][26][42][1024] bf16, 1-pixel zero halo for SAME 3x3 conv.
__global__ void pack_x_kernel(const float* __restrict__ x, __bf16* __restrict__ xpad) {
  size_t i = (size_t)blockIdx.x * blockDim.x + threadIdx.x;
  const size_t total = (size_t)BATCH * 26 * 42 * CIN;
  if (i >= total) return;
  int c = (int)(i & (CIN - 1));
  size_t p = i >> 10;
  int wq = (int)(p % 42); p /= 42;
  int hq = (int)(p % 26); p /= 26;
  int b  = (int)p;
  float v = 0.f;
  if (hq >= 1 && hq <= HF && wq >= 1 && wq <= WF)
    v = x[(((size_t)b * HF + (hq - 1)) * WF + (wq - 1)) * CIN + c];
  xpad[i] = f2bf(v);
}

// ---------------- kernel 2: weights HWIO(9216,512) -> bf16 [N][K] ----------------
__global__ void pack_w_kernel(const float* __restrict__ W, __bf16* __restrict__ wp) {
  size_t i = (size_t)blockIdx.x * blockDim.x + threadIdx.x;
  const size_t total = (size_t)KTOT * CMID;
  if (i >= total) return;
  int k = (int)(i % KTOT);
  int n = (int)(i / KTOT);
  wp[i] = f2bf(W[(size_t)k * CMID + n]);
}

// ---------------- kernel 3: implicit-GEMM 3x3 conv via WMMA bf16 ----------------
// C[M=15360, N=512] = A[M,K=9216] x W[K,N]; fused bias + ReLU.
// Block = 8 waves x 32 M rows = 256 M; all waves share one 64-wide N slice
// staged through double-buffered LDS via gfx1250 async global->LDS (ASYNCcnt).
// Each wave: 2 A fragments x 4 B fragments -> 8 WMMAs per 8 DS + 4 VMEM loads.
// sched_group_barrier pins the issue order loads(i+1) -> WMMAs(i) so waits
// are counted instead of full drains.
__global__ void __launch_bounds__(256) conv_gemm_kernel(
    const __bf16* __restrict__ xpad, const __bf16* __restrict__ wp,
    const float* __restrict__ bias, float* __restrict__ y) {
  __shared__ __align__(16) __bf16 ldsB[2][BCOLS * BSTRIDE];   // 2 x 17 KB

  const int tid   = threadIdx.x;
  const int lane  = tid & 31;
  const int wave  = tid >> 5;
  const int lrow  = lane & 15;   // M (for A) or N (for B/D) within 16x16 tile
  const int khalf = lane >> 4;   // K-half selector per ISA fragment layout
  const int m0    = blockIdx.x * MBLK + wave * 32;
  const int n0    = blockIdx.y * BCOLS;

  // A gather bases for the wave's two 16-row fragments
  const __bf16* xbase[2];
#pragma unroll
  for (int f = 0; f < 2; ++f) {
    const int m  = m0 + f * 16 + lrow;
    const int bi = m / PPI;
    const int hw = m - bi * PPI;
    const int hh = hw / WF;
    const int ww = hw - hh * WF;
    xbase[f] = xpad + (((size_t)(bi * 26 + hh)) * 42 + ww) * CIN + khalf * 8;
  }

  // B chunk copy: thread t moves 64B = row (t>>2), elems [(t&3)*32, +32)
  const int crow = tid >> 2;
  const int cq   = tid & 3;
  const __bf16* wsrc = wp + (size_t)(n0 + crow) * KTOT + cq * 32;
  __bf16* const ldst[2] = { &ldsB[0][crow * BSTRIDE + cq * 32],
                            &ldsB[1][crow * BSTRIDE + cq * 32] };

  union Frag { v16bf v; uint4 q[2]; };
  v8f acc[2][4] = {};
  Frag a_buf[2][2];              // [step parity][frag]

  // flattened K step s: kidx = s>>5 (conv tap), elem offset = 32*s + 39936*kh.
  // One-step overfetch at s==KSTEPS stays inside the workspace (never used).
  auto load_a = [&](int s, int f, Frag& a) {
    const int kh = (s >> 5) / 3;
    const __bf16* ap = xbase[f] + kh * (39 * 1024) + (s << 5);
    a.q[0] = *(const uint4*)(ap);
    a.q[1] = *(const uint4*)(ap + 16);
  };
  // 4 B fragments (64 N cols) for one K sub-step from the current LDS buffer
  auto load_b = [&](const __bf16* lbase, int sub, Frag* b) {
    const __bf16* l = lbase + sub * 32;
#pragma unroll
    for (int t = 0; t < 4; ++t) {
      b[t].q[0] = *(const uint4*)(l + t * 16 * BSTRIDE);
      b[t].q[1] = *(const uint4*)(l + t * 16 * BSTRIDE + 16);
    }
  };

  // prologue: chunk 0 -> buffer 0
  {
    uint4 t0 = *(const uint4*)(wsrc + 0);
    uint4 t1 = *(const uint4*)(wsrc + 8);
    uint4 t2 = *(const uint4*)(wsrc + 16);
    uint4 t3 = *(const uint4*)(wsrc + 24);
    *(uint4*)(ldst[0] + 0)  = t0;
    *(uint4*)(ldst[0] + 8)  = t1;
    *(uint4*)(ldst[0] + 16) = t2;
    *(uint4*)(ldst[0] + 24) = t3;
  }
  __syncthreads();
  load_a(0, 0, a_buf[0][0]);
  load_a(0, 1, a_buf[0][1]);

#pragma unroll 2                        // folds all (c&1) buffer selectors
  for (int c = 0; c < NCHUNK; ++c) {
    // issue next chunk's B fetch before consuming the current one
    // (final-chunk overfetch lands in the workspace and in the dead buffer)
    const __bf16* src = wsrc + (size_t)(c + 1) * CHUNK_K;
    __bf16* dst = ldst[(c + 1) & 1];
#if HAVE_ASYNC_LDS
    __builtin_amdgcn_global_load_async_to_lds_b128((gv4*)(src + 0),  (lv4*)(dst + 0),  0, 0);
    __builtin_amdgcn_global_load_async_to_lds_b128((gv4*)(src + 8),  (lv4*)(dst + 8),  0, 0);
    __builtin_amdgcn_global_load_async_to_lds_b128((gv4*)(src + 16), (lv4*)(dst + 16), 0, 0);
    __builtin_amdgcn_global_load_async_to_lds_b128((gv4*)(src + 24), (lv4*)(dst + 24), 0, 0);
#else
    uint4 p0 = *(const uint4*)(src + 0);
    uint4 p1 = *(const uint4*)(src + 8);
    uint4 p2 = *(const uint4*)(src + 16);
    uint4 p3 = *(const uint4*)(src + 24);
#endif

    const __bf16* lbase = &ldsB[c & 1][lrow * BSTRIDE + khalf * 8];
    Frag bf[2][4];
    load_b(lbase, 0, bf[0]);            // prime sub-step 0
    SGB(0x100, 8);                      // DS read x8 (prime)
#pragma unroll
    for (int sub = 0; sub < 4; ++sub) {
      const int s = c * 4 + sub;
      load_a(s + 1, 0, a_buf[(s + 1) & 1][0]);          // A one step ahead
      load_a(s + 1, 1, a_buf[(s + 1) & 1][1]);
      SGB(0x020, 4);                    // VMEM read x4 (next A)
      if (sub < 3) {
        load_b(lbase, sub + 1, bf[(sub + 1) & 1]);      // B one step ahead
        SGB(0x100, 8);                  // DS read x8 (next B)
      }
      const v16bf av0 = a_buf[s & 1][0].v;
      const v16bf av1 = a_buf[s & 1][1].v;
      const Frag* b = bf[sub & 1];
      acc[0][0] = __builtin_amdgcn_wmma_f32_16x16x32_bf16(false, av0, false, b[0].v, (short)0, acc[0][0], false, false);
      acc[0][1] = __builtin_amdgcn_wmma_f32_16x16x32_bf16(false, av0, false, b[1].v, (short)0, acc[0][1], false, false);
      acc[0][2] = __builtin_amdgcn_wmma_f32_16x16x32_bf16(false, av0, false, b[2].v, (short)0, acc[0][2], false, false);
      acc[0][3] = __builtin_amdgcn_wmma_f32_16x16x32_bf16(false, av0, false, b[3].v, (short)0, acc[0][3], false, false);
      acc[1][0] = __builtin_amdgcn_wmma_f32_16x16x32_bf16(false, av1, false, b[0].v, (short)0, acc[1][0], false, false);
      acc[1][1] = __builtin_amdgcn_wmma_f32_16x16x32_bf16(false, av1, false, b[1].v, (short)0, acc[1][1], false, false);
      acc[1][2] = __builtin_amdgcn_wmma_f32_16x16x32_bf16(false, av1, false, b[2].v, (short)0, acc[1][2], false, false);
      acc[1][3] = __builtin_amdgcn_wmma_f32_16x16x32_bf16(false, av1, false, b[3].v, (short)0, acc[1][3], false, false);
      SGB(0x008, 8);                    // WMMA x8
    }

#if HAVE_ASYNC_LDS
#if __has_builtin(__builtin_amdgcn_s_wait_asynccnt)
    __builtin_amdgcn_s_wait_asynccnt(0);
#else
    asm volatile("s_wait_asynccnt 0" ::: "memory");
#endif
#else
    *(uint4*)(dst + 0)  = p0;
    *(uint4*)(dst + 8)  = p1;
    *(uint4*)(dst + 16) = p2;
    *(uint4*)(dst + 24) = p3;
#endif
    __syncthreads();   // next buffer ready; previous buffer free for overwrite
  }

  // D layout: VGPR r -> M = m0 + f*16 + khalf*8 + r, N = n0 + t*16 + (lane&15)
#pragma unroll
  for (int f = 0; f < 2; ++f) {
#pragma unroll
    for (int t = 0; t < 4; ++t) {
      const int n_out = n0 + t * 16 + lrow;
      const float bv = bias[n_out];
#pragma unroll
      for (int r = 0; r < 8; ++r) {
        const int m_out = m0 + f * 16 + khalf * 8 + r;
        float v = acc[f][t][r] + bv;
        y[(size_t)m_out * CMID + n_out] = v > 0.f ? v : 0.f;
      }
    }
  }
}

// ---------------- kernel 4: 1x1 heads + softmax + anchor decode ----------------
__global__ void __launch_bounds__(64) heads_kernel(
    const float* __restrict__ y,
    const float* __restrict__ Wc, const float* __restrict__ bc,
    const float* __restrict__ Wr, const float* __restrict__ br,
    float* __restrict__ scores, float* __restrict__ boxes) {
  __shared__ float row[CMID];
  __shared__ float outv[64];
  const int m = blockIdx.x;
  const int tid = threadIdx.x;
  for (int c = tid; c < CMID; c += 64) row[c] = y[(size_t)m * CMID + c];
  __syncthreads();
  if (tid < 54) {
    float s;
    if (tid < 18) {
      s = bc[tid];
      for (int c = 0; c < CMID; ++c) s += row[c] * Wc[c * 18 + tid];
    } else {
      const int j = tid - 18;
      s = br[j];
      for (int c = 0; c < CMID; ++c) s += row[c] * Wr[c * 36 + j];
    }
    outv[tid] = s;
  }
  __syncthreads();
  if (tid < NA) {
    const int bi = m / PPI;
    const int hw = m - bi * PPI;
    const int hh = hw / WF, ww = hw - hh * WF;
    const float l0 = outv[2 * tid], l1 = outv[2 * tid + 1];
    const float sc = 1.f / (1.f + __expf(l0 - l1));     // softmax(...)[1]
    const float dx = outv[18 + 4 * tid + 0];
    const float dy = outv[18 + 4 * tid + 1];
    const float dw = outv[18 + 4 * tid + 2];
    const float dh = outv[18 + 4 * tid + 3];
    const float scl[3] = {8.f, 16.f, 32.f};
    const float rat[3] = {0.5f, 1.f, 2.f};
    const float s_ = scl[tid / 3], r_ = rat[tid - (tid / 3) * 3];
    const float sq = sqrtf(r_);
    const float ha = s_ * sq, wa = s_ / sq;
    const float xc = (ww + 0.5f) * 32.f, yc = (hh + 0.5f) * 32.f;
    const float px = dx * wa + xc, py = dy * ha + yc;
    const float pw = __expf(dw) * wa, ph = __expf(dh) * ha;
    float x1 = px - 0.5f * pw, y1 = py - 0.5f * ph;
    float x2 = px + 0.5f * pw, y2 = py + 0.5f * ph;
    x1 = fminf(fmaxf(x1, 0.f), 1279.f);
    y1 = fminf(fmaxf(y1, 0.f), 767.f);
    x2 = fminf(fmaxf(x2, 0.f), 1279.f);
    y2 = fminf(fmaxf(y2, 0.f), 767.f);
    const int idx = hw * NA + tid;
    scores[(size_t)bi * NANCH + idx] = sc;
    float* bx = boxes + ((size_t)bi * NANCH + idx) * 4;
    bx[0] = x1; bx[1] = y1; bx[2] = x2; bx[3] = y2;
  }
}

// ---------------- kernel 5: top-6000 gate + 300-step NMS (1 block / image) ----------------
__global__ void __launch_bounds__(256) nms_kernel(
    const float* __restrict__ scores, const float* __restrict__ boxes,
    float* __restrict__ out) {
  __shared__ float sc[NANCH];      // 34.6 KB of the 320 KB WGP LDS
  __shared__ float rv[256];
  __shared__ int   ri[256];
  __shared__ int   rc[256];
  const int bi = blockIdx.x;
  const int tid = threadIdx.x;
  const float* bb = boxes + (size_t)bi * NANCH * 4;
  for (int i = tid; i < NANCH; i += 256) sc[i] = scores[(size_t)bi * NANCH + i];
  __syncthreads();

  // Binary-search the 6000th-largest score; scores are softmax outputs in (0,1).
  float lo = 0.f, hi = 1.f;
  for (int it = 0; it < 24; ++it) {
    const float mid = 0.5f * (lo + hi);
    int cnt = 0;
    for (int i = tid; i < NANCH; i += 256) cnt += (sc[i] > mid);
    rc[tid] = cnt;
    __syncthreads();
    for (int s = 128; s > 0; s >>= 1) {
      if (tid < s) rc[tid] += rc[tid + s];
      __syncthreads();
    }
    const int total = rc[0];
    __syncthreads();
    if (total >= PRE_TOPK) lo = mid; else hi = mid;
  }
  for (int i = tid; i < NANCH; i += 256) if (sc[i] <= lo) sc[i] = -1.f;
  __syncthreads();

  for (int it = 0; it < POST_N; ++it) {
    // block-wide argmax
    float bv = -1e30f; int bj = 0;
    for (int i = tid; i < NANCH; i += 256) {
      const float v = sc[i];
      if (v > bv) { bv = v; bj = i; }
    }
    rv[tid] = bv; ri[tid] = bj;
    __syncthreads();
    for (int s = 128; s > 0; s >>= 1) {
      if (tid < s && rv[tid + s] > rv[tid]) { rv[tid] = rv[tid + s]; ri[tid] = ri[tid + s]; }
      __syncthreads();
    }
    const int j = ri[0];
    const float sj = rv[0];
    __syncthreads();
    const bool valid = sj > -0.5f;
    if (tid == 0) {
      float* o = out + ((size_t)bi * POST_N + it) * 4;
      if (valid) { o[0] = bb[j*4+0]; o[1] = bb[j*4+1]; o[2] = bb[j*4+2]; o[3] = bb[j*4+3]; }
      else       { o[0] = 0.f; o[1] = 0.f; o[2] = 0.f; o[3] = 0.f; }
    }
    if (valid) {
      const float jx1 = bb[j*4+0], jy1 = bb[j*4+1], jx2 = bb[j*4+2], jy2 = bb[j*4+3];
      const float ja = (jx2 - jx1) * (jy2 - jy1);
      for (int i = tid; i < NANCH; i += 256) {
        const float x1 = bb[i*4+0], y1 = bb[i*4+1], x2 = bb[i*4+2], y2 = bb[i*4+3];
        const float xx1 = fmaxf(jx1, x1), yy1 = fmaxf(jy1, y1);
        const float xx2 = fminf(jx2, x2), yy2 = fminf(jy2, y2);
        const float inter = fmaxf(xx2 - xx1, 0.f) * fmaxf(yy2 - yy1, 0.f);
        const float ar = (x2 - x1) * (y2 - y1);
        const float iou = inter / (ja + ar - inter + 1e-8f);
        if (iou > NMS_T || i == j) sc[i] = -1.f;
      }
    }
    __syncthreads();
  }
}

// ---------------- launch ----------------
extern "C" void kernel_launch(void* const* d_in, const int* in_sizes, int n_in,
                              void* d_out, int out_size, void* d_ws, size_t ws_size,
                              hipStream_t stream) {
  const float* feat = (const float*)d_in[0];
  const float* Wsh  = (const float*)d_in[1];
  const float* bsh  = (const float*)d_in[2];
  const float* Wc   = (const float*)d_in[3];
  const float* bc   = (const float*)d_in[4];
  const float* Wr   = (const float*)d_in[5];
  const float* br   = (const float*)d_in[6];
  float* out = (float*)d_out;
  char* ws = (char*)d_ws;

  // workspace layout (~80 MB)
  const size_t sz_xpad = (size_t)BATCH * 26 * 42 * CIN * 2;  // 35.8 MB bf16
  const size_t sz_wp   = (size_t)KTOT * CMID * 2;            //  9.4 MB bf16
  const size_t sz_y    = (size_t)MTOT * CMID * 4;            // 31.5 MB f32
  const size_t sz_sc   = (size_t)BATCH * NANCH * 4;          //  0.6 MB
  const size_t off_xpad = 0;
  const size_t off_wp   = off_xpad + sz_xpad;
  const size_t off_y    = off_wp + sz_wp;
  const size_t off_sc   = off_y + sz_y;
  const size_t off_bx   = off_sc + sz_sc;

  __bf16* xpad = (__bf16*)(ws + off_xpad);
  __bf16* wp   = (__bf16*)(ws + off_wp);
  float*  y    = (float*)(ws + off_y);
  float*  sc   = (float*)(ws + off_sc);
  float*  bx   = (float*)(ws + off_bx);

  {
    const size_t n = (size_t)BATCH * 26 * 42 * CIN;
    pack_x_kernel<<<(unsigned)((n + 255) / 256), 256, 0, stream>>>(feat, xpad);
  }
  {
    const size_t n = (size_t)KTOT * CMID;
    pack_w_kernel<<<(unsigned)((n + 255) / 256), 256, 0, stream>>>(Wsh, wp);
  }
  dim3 cgrid(MTOT / MBLK, CMID / BCOLS);   // 60 x 8
  conv_gemm_kernel<<<cgrid, 256, 0, stream>>>(xpad, wp, bsh, y);
  heads_kernel<<<MTOT, 64, 0, stream>>>(y, Wc, bc, Wr, br, sc, bx);
  nms_kernel<<<BATCH, 256, 0, stream>>>(sc, bx, out);
}
